// Attention_4569845203205
// MI455X (gfx1250) — matrix operand order repaired and checked by reading, not compiled, not run on hardware
//
#include <hip/hip_runtime.h>

// ---------------------------------------------------------------------------
// MI455X (gfx1250) fused non-local attention block.
// Dominant cost: 2 * (2*B*N^2*Co) ~= 8.6 GFLOP -> bf16 WMMA w/ f32 accumulate.
// Flash-style fusion: g[N][N] never materialized; S^T register tiles from
// v_wmma_f32_16x16x32_bf16 feed directly (lane-local relu+pack) into the
// second WMMA.  Phase-2 m-loop split over 2 partitions for occupancy
// (256 WGs / 1024 waves); LDS tiles double-buffered and filled with
// GLOBAL_LOAD_ASYNC_TO_LDS_B128 (ASYNCcnt) when available.
// ---------------------------------------------------------------------------

typedef __attribute__((ext_vector_type(16))) __bf16 v16bf;
typedef __attribute__((ext_vector_type(8)))  float  v8f;

union UBF { v16bf v; __bf16 e[16]; uint4 q[2]; };
union UF8 { v8f  v; float  e[8]; };

static constexpr int Bb    = 2;     // batch
static constexpr int CI    = 128;   // input channels
static constexpr int CO    = 64;    // projection channels
static constexpr int NN    = 4096;  // T*H*W
static constexpr int NPART = 2;     // m-loop partitions in phase 2

__device__ __forceinline__ __bf16 f2bf(float f) { return (__bf16)f; }

__device__ __forceinline__ v8f wmma_bf16(v16bf a, v16bf b, v8f c) {
  // (neg_a, A, neg_b, B, c_mod, C, reuse_a, reuse_b)
  return __builtin_amdgcn_wmma_f32_16x16x32_bf16(false, a, false, b, (short)0, c,
                                                 false, false);
}

// ------------------------- async global->LDS support -----------------------
#if defined(__has_builtin)
#if __has_builtin(__builtin_amdgcn_global_load_async_to_lds_b128)
#define HAVE_ASYNC 1
#endif
#endif
#ifndef HAVE_ASYNC
#define HAVE_ASYNC 0
#endif

#if HAVE_ASYNC
// Builtin signature (from clang diagnostic): (int4 AS1*, int4 AS3*, Ii, Ii)
typedef int v4i_t __attribute__((vector_size(16)));
typedef __attribute__((address_space(1))) v4i_t gbl_v4i_t;
typedef __attribute__((address_space(3))) v4i_t lds_v4i_t;
__device__ __forceinline__ void cp_async_b128(const void* g, void* l) {
  __builtin_amdgcn_global_load_async_to_lds_b128(
      (gbl_v4i_t*)g, (lds_v4i_t*)l, /*offset=*/0, /*cpol=*/0);
}
__device__ __forceinline__ void wait_async() {
#if __has_builtin(__builtin_amdgcn_s_wait_asynccnt)
  __builtin_amdgcn_s_wait_asynccnt(0);
#else
  asm volatile("s_wait_asynccnt 0x0" ::: "memory");
#endif
}
#else
__device__ __forceinline__ void wait_async() {}
#endif

// ---------------------------------------------------------------------------
// Phase 1: theta/phi/psi = W @ z + b.  Per 16-n tile one wave does
// 3 proj * 4 co-tiles * 4 K-steps = 48 WMMAs.  Outputs (bf16):
//   phiT [B][N][64]  (B-operand source, lane=col n, contiguous c)
//   psiT [B][N][64]  (A-operand source, lane=row m, contiguous c)
//   theta[B][64][N]  (B-operand source for 2nd matmul, lane=col co, contig m)
// ---------------------------------------------------------------------------
__global__ __launch_bounds__(128)
void proj_kernel(const float* __restrict__ z,
                 const float* __restrict__ w_theta, const float* __restrict__ b_theta,
                 const float* __restrict__ w_phi,   const float* __restrict__ b_phi,
                 const float* __restrict__ w_psi,   const float* __restrict__ b_psi,
                 __bf16* __restrict__ phiT, __bf16* __restrict__ psiT,
                 __bf16* __restrict__ theta)
{
  const int wave = threadIdx.x >> 5;
  const int lane = threadIdx.x & 31;
  const int lh   = lane & 15;
  const int hi   = lane >> 4;
  const int nblk = NN / 64;
  const int b    = blockIdx.x / nblk;
  const int n0   = (blockIdx.x % nblk) * 64 + wave * 16;
  const int n    = n0 + lh;

  // B operands: z column n as bf16.  4 K-steps of 32 channels.
  // Contiguous-k B order used here (A matches): lanes 0-15 K = c0..c0+15, lanes 16-31 K = c0+16..c0+31.
  UBF bz[4];
  const float* zp = z + ((size_t)b * CI) * NN + n;
#pragma unroll
  for (int s = 0; s < 4; ++s) {
    const int c0 = s * 32 + hi * 16;
#pragma unroll
    for (int j = 0; j < 16; ++j)
      bz[s].e[j] = f2bf(zp[(size_t)(c0 + j) * NN]);
  }

  const float* Ws[3]   = { w_theta, w_phi, w_psi };
  const float* Bias[3] = { b_theta, b_phi, b_psi };

#pragma unroll
  for (int p = 0; p < 3; ++p) {
    const float* w    = Ws[p];
    const float* bias = Bias[p];
#pragma unroll
    for (int t = 0; t < 4; ++t) {
      const int co0 = t * 16;
      UF8 acc;
#pragma unroll
      for (int r = 0; r < 8; ++r) acc.e[r] = 0.0f;
#pragma unroll
      for (int s = 0; s < 4; ++s) {
        const int c0 = s * 32;
        // Contiguous-k A order (matches B): lane row = co0+lh; lanes<16 K=c0..c0+15,
        // lanes>=16 K=c0+16..c0+31.
        UBF aw;
        const float* wrow = w + (co0 + lh) * CI;
#pragma unroll
        for (int j = 0; j < 8; ++j) {
          aw.e[j]     = f2bf(wrow[c0 + hi * 16 + j]);
          aw.e[j + 8] = f2bf(wrow[c0 + hi * 16 + 8 + j]);
        }
        acc.v = wmma_bf16(aw.v, bz[s].v, acc.v);
      }
      // D tile: lane col = n, VGPR r -> row co = co0 + hi*8 + r
      if (p == 0) {
#pragma unroll
        for (int r = 0; r < 8; ++r) {
          const int co = co0 + hi * 8 + r;
          theta[((size_t)b * CO + co) * NN + n] = f2bf(acc.e[r] + bias[co]);
        }
      } else {
        __bf16* outT = (p == 1) ? phiT : psiT;
        __bf16* dst  = outT + ((size_t)b * NN + n) * CO + co0 + hi * 8;
#pragma unroll
        for (int r = 0; r < 8; ++r)
          dst[r] = f2bf(acc.e[r] + bias[co0 + hi * 8 + r]);
      }
    }
  }
}

// ---------------------------------------------------------------------------
// Phase 2: tmpP[part][n][co] = sum_{m in part} relu((phi^T psi)[n][m]/N) * theta[co][m]
// WG = 128 threads (4 waves; each wave a 16-row n tile; 64 n rows per WG).
// Each WG handles NN/NPART m values in steps of 32; psiT/theta tiles staged in
// double-buffered LDS (async global->LDS when available) -> ONE barrier/iter.
// 8 WMMAs per m-step per wave.
// ---------------------------------------------------------------------------
__global__ __launch_bounds__(128)
void attn_kernel(const __bf16* __restrict__ phiT, const __bf16* __restrict__ psiT,
                 const __bf16* __restrict__ theta, float* __restrict__ tmpP)
{
  __shared__ __bf16 psi_s[2][32][72];  // [buf][m'][c], 144B rows (16B aligned)
  __shared__ __bf16 th_s[2][64][40];   // [buf][co][m'], 80B rows (16B aligned)

  const int wave = threadIdx.x >> 5;
  const int lane = threadIdx.x & 31;
  const int lh   = lane & 15;
  const int hi   = lane >> 4;
  const int nblk = NN / 64;

  int id = blockIdx.x;
  const int part = id / (Bb * nblk);
  id %= (Bb * nblk);
  const int b  = id / nblk;
  const int n0 = (id % nblk) * 64 + wave * 16;
  const int n  = n0 + lh;

  const int mbeg  = part * (NN / NPART);
  const int ITERS = (NN / NPART) / 32;

  // Stage one 32-m tile of psiT (32x64) and theta (64x32) into LDS buffer.
  const int sr  = threadIdx.x >> 2;         // psi row 0..31
  const int scs = (threadIdx.x & 3) * 16;   // psi col seg
  const int sco = threadIdx.x >> 1;         // theta row 0..63
  const int sms = (threadIdx.x & 1) * 16;   // theta col seg
  auto stage = [&](int m0, int buf) {
    const __bf16* srcp = psiT  + ((size_t)b * NN + m0 + sr) * CO + scs;
    const __bf16* srct = theta + ((size_t)b * CO + sco) * NN + m0 + sms;
#if HAVE_ASYNC
    cp_async_b128(srcp,     &psi_s[buf][sr][scs]);
    cp_async_b128(srcp + 8, &psi_s[buf][sr][scs + 8]);
    cp_async_b128(srct,     &th_s[buf][sco][sms]);
    cp_async_b128(srct + 8, &th_s[buf][sco][sms + 8]);
#else
    uint4 p0 = *(const uint4*)(srcp);
    uint4 p1 = *(const uint4*)(srcp + 8);
    uint4 t0 = *(const uint4*)(srct);
    uint4 t1 = *(const uint4*)(srct + 8);
    *(uint4*)&psi_s[buf][sr][scs]      = p0;
    *(uint4*)&psi_s[buf][sr][scs + 8]  = p1;
    *(uint4*)&th_s[buf][sco][sms]      = t0;
    *(uint4*)&th_s[buf][sco][sms + 8]  = t1;
#endif
  };

  // phi B operands (fixed per wave): lane col = n; contiguous c in phiT.
  UBF bphi[2];
  {
    const __bf16* pr = phiT + ((size_t)b * NN + n) * CO;
#pragma unroll
    for (int s = 0; s < 2; ++s) {
      const __bf16* src = pr + s * 32 + hi * 8;
      bphi[s].q[0] = *(const uint4*)(src);
      bphi[s].q[1] = *(const uint4*)(src + 16);
    }
  }

  UF8 acc[4];
#pragma unroll
  for (int t = 0; t < 4; ++t)
#pragma unroll
    for (int r = 0; r < 8; ++r) acc[t].e[r] = 0.0f;

  const float invN = 1.0f / (float)NN;

  stage(mbeg, 0);  // prologue fill of buffer 0

  for (int it = 0; it < ITERS; ++it) {
    const int buf = it & 1;
    wait_async();     // this wave's async fills of buf complete
    __syncthreads();  // all waves' fills of buf visible; prior reads of buf^1 done
    if (it + 1 < ITERS) stage(mbeg + (it + 1) * 32, buf ^ 1);  // overlap w/ compute

    // S^T[m'][n] = psi^T * phi  (two 16-m tiles, K = c = 64 -> 2 steps)
    UF8 st0, st1;
#pragma unroll
    for (int r = 0; r < 8; ++r) { st0.e[r] = 0.0f; st1.e[r] = 0.0f; }
#pragma unroll
    for (int s = 0; s < 2; ++s) {
      const int c0 = s * 32;
      UBF a0, a1;   // A rows = m'; lane row = lh (tile0) / 16+lh (tile1)
      a0.q[0] = *(const uint4*)&psi_s[buf][lh][c0 + hi * 8];
      a0.q[1] = *(const uint4*)&psi_s[buf][lh][c0 + 16 + hi * 8];
      a1.q[0] = *(const uint4*)&psi_s[buf][16 + lh][c0 + hi * 8];
      a1.q[1] = *(const uint4*)&psi_s[buf][16 + lh][c0 + 16 + hi * 8];
      st0.v = wmma_bf16(a0.v, bphi[s].v, st0.v);
      st1.v = wmma_bf16(a1.v, bphi[s].v, st1.v);
    }

    // Lane-local: D layout of st0/st1 == A-operand layout of 2nd matmul.
    UBF a2;
#pragma unroll
    for (int j = 0; j < 8; ++j) {
      const float v0 = st0.e[j] * invN;
      const float v1 = st1.e[j] * invN;
      a2.e[j]     = f2bf(v0 > 0.0f ? v0 : 0.0f);
      a2.e[j + 8] = f2bf(v1 > 0.0f ? v1 : 0.0f);
    }

    // acc[n][co] += relu(S) * theta^T   (4 co tiles)
#pragma unroll
    for (int t = 0; t < 4; ++t) {
      UBF bt;   // B[k=m'][col=co]; lane col = t*16+lh; K half by hi
      bt.q[0] = *(const uint4*)&th_s[buf][t * 16 + lh][hi * 8];
      bt.q[1] = *(const uint4*)&th_s[buf][t * 16 + lh][16 + hi * 8];
      acc[t].v = wmma_bf16(a2.v, bt.v, acc[t].v);
    }
  }

  // Store fp32 partial: tmpP[part][b][n][co].
  // D: lane col co = t*16+lh, VGPR r -> n = n0 + hi*8 + r.
  float* dst = tmpP + (((size_t)part * Bb + b) * NN) * CO;
#pragma unroll
  for (int t = 0; t < 4; ++t)
#pragma unroll
    for (int r = 0; r < 8; ++r) {
      const int nn = n0 + hi * 8 + r;
      dst[(size_t)nn * CO + t * 16 + lh] = acc[t].e[r];
    }
}

// ---------------------------------------------------------------------------
// Phase 3: out = w_v @ (sum of tmp partials) + b_v + z.  K = Co = 64.
// ---------------------------------------------------------------------------
__global__ __launch_bounds__(128)
void out_kernel(const float* __restrict__ z, const float* __restrict__ w_v,
                const float* __restrict__ b_v, const float* __restrict__ tmpP,
                float* __restrict__ out)
{
  const int wave = threadIdx.x >> 5;
  const int lane = threadIdx.x & 31;
  const int lh   = lane & 15;
  const int hi   = lane >> 4;
  const int nblk = NN / 64;
  const int b    = blockIdx.x / nblk;
  const int n0   = (blockIdx.x % nblk) * 64 + wave * 16;
  const int n    = n0 + lh;

  // B[k=co][col=n] from summed fp32 partials (contiguous per lane).
  UBF btmp[2];
  {
    const float* t0 = tmpP + (((size_t)0 * Bb + b) * NN + n) * CO;
    const float* t1 = tmpP + (((size_t)1 * Bb + b) * NN + n) * CO;
#pragma unroll
    for (int s = 0; s < 2; ++s) {
      const int base = s * 32 + hi * 16;
#pragma unroll
      for (int j = 0; j < 16; ++j)
        btmp[s].e[j] = f2bf(t0[base + j] + t1[base + j]);
    }
  }

#pragma unroll
  for (int t = 0; t < 8; ++t) {
    const int ci0 = t * 16;
    UF8 acc;
#pragma unroll
    for (int r = 0; r < 8; ++r) acc.e[r] = 0.0f;
#pragma unroll
    for (int s = 0; s < 2; ++s) {
      const int c0 = s * 32;
      UBF aw;  // A rows = ci; w_v is [CI][CO] row-major -> contiguous K
      const float* wrow = w_v + (ci0 + lh) * CO;
#pragma unroll
      for (int j = 0; j < 8; ++j) {
        aw.e[j]     = f2bf(wrow[c0 + hi * 16 + j]);
        aw.e[j + 8] = f2bf(wrow[c0 + hi * 16 + 8 + j]);
      }
      acc.v = wmma_bf16(aw.v, btmp[s].v, acc.v);
    }
    // D: lane col = n, VGPR r -> ci = ci0 + hi*8 + r.  Residual add + bias.
#pragma unroll
    for (int r = 0; r < 8; ++r) {
      const int ci = ci0 + hi * 8 + r;
      const size_t idx = ((size_t)b * CI + ci) * NN + n;
      out[idx] = acc.e[r] + b_v[ci] + z[idx];
    }
  }
}

// ---------------------------------------------------------------------------
extern "C" void kernel_launch(void* const* d_in, const int* in_sizes, int n_in,
                              void* d_out, int out_size, void* d_ws, size_t ws_size,
                              hipStream_t stream) {
  (void)in_sizes; (void)n_in; (void)out_size; (void)ws_size;
  const float* z       = (const float*)d_in[0];
  const float* w_theta = (const float*)d_in[1];
  const float* b_theta = (const float*)d_in[2];
  const float* w_phi   = (const float*)d_in[3];
  const float* b_phi   = (const float*)d_in[4];
  const float* w_psi   = (const float*)d_in[5];
  const float* b_psi   = (const float*)d_in[6];
  const float* w_v     = (const float*)d_in[7];
  const float* b_v     = (const float*)d_in[8];
  float* out = (float*)d_out;

  const size_t planeB = (size_t)Bb * NN * CO * sizeof(__bf16);  // 1 MB each
  char* ws = (char*)d_ws;
  __bf16* phiT  = (__bf16*)(ws);
  __bf16* psiT  = (__bf16*)(ws + planeB);
  __bf16* theta = (__bf16*)(ws + 2 * planeB);
  float*  tmpP  = (float*)(ws + 3 * planeB);  // NPART*B*N*CO fp32 = 4 MB

  const dim3 block(128);  // 4 waves (wave32)
  proj_kernel<<<dim3(Bb * (NN / 64)), block, 0, stream>>>(
      z, w_theta, b_theta, w_phi, b_phi, w_psi, b_psi, phiT, psiT, theta);
  attn_kernel<<<dim3(Bb * (NN / 64) * NPART), block, 0, stream>>>(
      phiT, psiT, theta, tmpP);
  out_kernel<<<dim3(Bb * (NN / 64)), block, 0, stream>>>(
      z, w_v, b_v, tmpP, out);
}
